// Attention_76622216560757
// MI455X (gfx1250) — matrix operand order repaired and checked by reading, not compiled, not run on hardware
//
#include <hip/hip_runtime.h>
#include <hip/hip_bf16.h>

#define S_LEN 2048
#define HID_DIM 2560
#define NH 32
#define NKV 8
#define HD 128
#define QSTRIDE (NH * HD)    /* 4096 */
#define KVSTRIDE (NKV * HD)  /* 1024 */
#define SOFTMAX_SCALE 0.08838834764831845f

typedef __attribute__((ext_vector_type(16))) _Float16 v16h;
typedef __attribute__((ext_vector_type(8)))  _Float16 h8;
typedef __attribute__((ext_vector_type(4)))  _Float16 h4;
typedef __attribute__((ext_vector_type(8)))  float    v8f;
typedef __attribute__((ext_vector_type(4)))  float    f32x4;
typedef __attribute__((ext_vector_type(4)))  unsigned int u32x4;
typedef __attribute__((ext_vector_type(8)))  int      i32x8;
typedef __attribute__((ext_vector_type(4)))  int      i32x4;

__device__ __forceinline__ v8f wmma_f16(v16h a, v16h b, v8f c) {
  // D = A(16x32 f16) * B(32x16 f16) + C(16x16 f32)
  return __builtin_amdgcn_wmma_f32_16x16x32_f16(
      /*neg_a=*/false, a, /*neg_b=*/false, b,
      /*c_mod=*/(short)0, c, /*reuse_a=*/false, /*reuse_b=*/false);
}

// ---------------- Tensor Data Mover: 2D tile -> LDS ----------------
// D# per CDNA5 ISA 8.3/8.4: group0 = {count|flags, lds_addr, global_addr,
// type=2}; group1 = {data_size=2B, tensor dims, tile dims, dim0 stride}.
// tile: tile_d0 (elements, innermost/contiguous) x tile_d1 rows, packed
// contiguously into LDS at lds_off. Uses the 6-arg clang-23 builtin form.
__device__ __forceinline__ void tdm_load_2d(unsigned lds_off, const void* gptr,
                                            unsigned tensor_d0, unsigned tensor_d1,
                                            unsigned tile_d0, unsigned tile_d1,
                                            unsigned stride0_elems) {
  const unsigned long long ga = (unsigned long long)gptr;
  u32x4 g0;
  g0[0] = 1u;                                            // count=1 (valid D#)
  g0[1] = lds_off;                                       // lds_addr (bytes)
  g0[2] = (unsigned)(ga & 0xffffffffu);                  // global_addr[31:0]
  g0[3] = (unsigned)((ga >> 32) & 0x01ffffffu) | (2u << 30);  // [56:32], type=2
  i32x8 g1;
  g1[0] = (int)(1u << 16);                               // data_size=1 -> 2 bytes
  g1[1] = (int)((tensor_d0 & 0xffffu) << 16);            // tensor_dim0 lo
  g1[2] = (int)(((tensor_d0 >> 16) & 0xffffu) |
                ((tensor_d1 & 0xffffu) << 16));          // dim0 hi | dim1 lo
  g1[3] = (int)(((tensor_d1 >> 16) & 0xffffu) |
                ((tile_d0 & 0xffffu) << 16));            // dim1 hi | tile_dim0
  g1[4] = (int)(tile_d1 & 0xffffu);                      // tile_dim1 (tile_dim2=0)
  g1[5] = (int)stride0_elems;                            // tensor_dim0_stride lo
  g1[6] = 0;                                             // stride0 hi | stride1 lo
  g1[7] = 0;                                             // stride1 hi
  const i32x4 z4 = {};
  const i32x8 z8 = {};
  __builtin_amdgcn_tensor_load_to_lds(g0, g1, z4, z4, z8, 0);
}

// ---------------- LDS fragment readers (tile stored [row][32] f16) ---------
__device__ __forceinline__ v16h lds_a_frag(const _Float16* buf, int mbase) {
  const int lane = threadIdx.x & 31;
  const int hf = lane >> 4;
  const int m  = mbase + (lane & 15);
  const _Float16* p = buf + m * 32 + hf * 8;
  h8 x0 = *(const h8*)(p);
  h8 x1 = *(const h8*)(p + 16);
  v16h a;
#pragma unroll
  for (int j = 0; j < 8; ++j) { a[j] = x0[j]; a[8 + j] = x1[j]; }
  return a;
}

__device__ __forceinline__ v16h lds_b_frag(const _Float16* buf, int nbase) {
  const int lane = threadIdx.x & 31;
  const int hf = lane >> 4;
  const int n  = nbase + (lane & 15);
  const _Float16* p = buf + n * 32 + hf * 8;
  h8 x0 = *(const h8*)(p);
  h8 x1 = *(const h8*)(p + 16);
  v16h b;
#pragma unroll
  for (int j = 0; j < 8; ++j) { b[j] = x0[j]; b[8 + j] = x1[j]; }
  return b;
}

// ---------------- Global fragment readers (for flash attention) ------------
__device__ __forceinline__ v16h load_a16(const _Float16* __restrict__ base,
                                         int ld, int k0) {
  const int lane = threadIdx.x & 31;
  const int hf = lane >> 4;
  const int m  = lane & 15;
  const _Float16* p = base + (size_t)m * ld + k0 + hf * 8;
  h8 x0 = *(const h8*)(p);
  h8 x1 = *(const h8*)(p + 16);
  v16h a;
#pragma unroll
  for (int j = 0; j < 8; ++j) { a[j] = x0[j]; a[8 + j] = x1[j]; }
  return a;
}

__device__ __forceinline__ v16h load_b16(const _Float16* __restrict__ base,
                                         int ld, int n0, int k0) {
  const int lane = threadIdx.x & 31;
  const int hf = lane >> 4;
  const int n  = n0 + (lane & 15);
  const _Float16* p = base + (size_t)n * ld + k0 + hf * 8;
  h8 x0 = *(const h8*)(p);
  h8 x1 = *(const h8*)(p + 16);
  v16h b;
#pragma unroll
  for (int j = 0; j < 8; ++j) { b[j] = x0[j]; b[8 + j] = x1[j]; }
  return b;
}

// Elementwise f32 -> f16 (n multiple of 4).
__global__ void __launch_bounds__(256)
cvt_f16_kernel(const float* __restrict__ in, _Float16* __restrict__ out, int n) {
  const int i = (blockIdx.x * 256 + threadIdx.x) * 4;
  if (i >= n) return;
  f32x4 x = *(const f32x4*)(in + i);
  h4 y;
#pragma unroll
  for (int j = 0; j < 4; ++j) y[j] = (_Float16)x[j];
  *(h4*)(out + i) = y;
}

// C[M,N] = X[M,K] * W[N,K]^T, f16 operands staged to LDS by the Tensor Data
// Mover, double-buffered so the next slab's DMA overlaps this slab's WMMAs.
// Block = 128 threads = 4 waves (2x2); wave tile 64x64; block tile 128x128.
template <typename OutT>
__global__ void __launch_bounds__(128)
gemm16_tdm_kernel(const _Float16* __restrict__ X, const _Float16* __restrict__ W,
                  OutT* __restrict__ C, int N, int K) {
  __shared__ __align__(64) _Float16 Abuf[2][128 * 32];   // 8KB each
  __shared__ __align__(64) _Float16 Bbuf[2][128 * 32];

  const int wave = threadIdx.x >> 5;
  const int wm = wave >> 1, wn = wave & 1;
  const int mblk = blockIdx.y * 128;
  const int nblk = blockIdx.x * 128;

  // prologue: wave0 DMAs the first A/B slabs
  if (wave == 0) {
    tdm_load_2d((unsigned)(unsigned long long)(void*)&Abuf[0][0],
                X + (size_t)mblk * K, K, 0x7fffffffu, 32, 128, K);
    tdm_load_2d((unsigned)(unsigned long long)(void*)&Bbuf[0][0],
                W + (size_t)nblk * K, K, 0x7fffffffu, 32, 128, K);
    __builtin_amdgcn_s_wait_tensorcnt(0);
  }
  __syncthreads();

  v8f acc[4][4] = {};
  int cur = 0;
  for (int k0 = 0; k0 < K; k0 += 32) {
    // kick off next slab's DMA while we compute on the current one
    if (wave == 0 && (k0 + 32) < K) {
      tdm_load_2d((unsigned)(unsigned long long)(void*)&Abuf[1 - cur][0],
                  X + (size_t)mblk * K + k0 + 32, K, 0x7fffffffu, 32, 128, K);
      tdm_load_2d((unsigned)(unsigned long long)(void*)&Bbuf[1 - cur][0],
                  W + (size_t)nblk * K + k0 + 32, K, 0x7fffffffu, 32, 128, K);
    }

    v16h a[4], b[4];
#pragma unroll
    for (int i = 0; i < 4; ++i) a[i] = lds_a_frag(&Abuf[cur][0], wm * 64 + i * 16);
#pragma unroll
    for (int j = 0; j < 4; ++j) b[j] = lds_b_frag(&Bbuf[cur][0], wn * 64 + j * 16);
#pragma unroll
    for (int i = 0; i < 4; ++i)
#pragma unroll
      for (int j = 0; j < 4; ++j)
        acc[i][j] = wmma_f16(a[i], b[j], acc[i][j]);

    if (wave == 0) __builtin_amdgcn_s_wait_tensorcnt(0);
    __syncthreads();
    cur ^= 1;
  }

  const int lane = threadIdx.x & 31;
  const int hf = lane >> 4, nn = lane & 15;
#pragma unroll
  for (int i = 0; i < 4; ++i)
#pragma unroll
    for (int j = 0; j < 4; ++j)
#pragma unroll
      for (int r = 0; r < 8; ++r)
        C[(size_t)(mblk + wm * 64 + i * 16 + hf * 8 + r) * N +
          (nblk + wn * 64 + j * 16 + nn)] = (OutT)acc[i][j][r];
}

// Per-(s,head) RMSNorm + RoPE, in place on f16 [S, nheads*128], f32 math.
// One wave per row; lane owns d, d+32, d+64, d+96 so rotate_half is intra-lane.
__global__ void __launch_bounds__(32)
norm_rope16_kernel(_Float16* __restrict__ X, const float* __restrict__ freqs,
                   const float* __restrict__ w, int rowstride, float outscale) {
  const int s = blockIdx.x;
  const int h = blockIdx.y;
  _Float16* row = X + (size_t)s * rowstride + h * HD;
  const int lane = threadIdx.x;

  float x0 = (float)row[lane];
  float x1 = (float)row[lane + 32];
  float x2 = (float)row[lane + 64];
  float x3 = (float)row[lane + 96];
  float ss = x0 * x0 + x1 * x1 + x2 * x2 + x3 * x3;
#pragma unroll
  for (int msk = 1; msk < 32; msk <<= 1) ss += __shfl_xor(ss, msk, 32);
  const float rinv = rsqrtf(ss * (1.0f / 128.0f) + 1e-6f);

  const float n0 = x0 * rinv * w[lane];
  const float n1 = x1 * rinv * w[lane + 32];
  const float n2 = x2 * rinv * w[lane + 64];
  const float n3 = x3 * rinv * w[lane + 96];

  const float* cosr = freqs + (size_t)s * HD;
  const float* sinr = freqs + (size_t)S_LEN * HD + (size_t)s * HD;
  row[lane]      = (_Float16)((n0 * cosr[lane]      - n2 * sinr[lane])      * outscale);
  row[lane + 32] = (_Float16)((n1 * cosr[lane + 32] - n3 * sinr[lane + 32]) * outscale);
  row[lane + 64] = (_Float16)((n2 * cosr[lane + 64] + n0 * sinr[lane + 64]) * outscale);
  row[lane + 96] = (_Float16)((n3 * cosr[lane + 96] + n1 * sinr[lane + 96]) * outscale);
}

// V [S, HKV*D] f16 -> V^T [HKV][D][S] f16 so the P*V B-fragment is contiguous.
__global__ void __launch_bounds__(256)
vtrans_kernel(const _Float16* __restrict__ v, _Float16* __restrict__ vt) {
  const int c = blockIdx.y;                       // h*128 + d
  const int s = blockIdx.x * 256 + threadIdx.x;
  vt[(size_t)c * S_LEN + s] = v[(size_t)s * KVSTRIDE + c];
}

// Flash attention: one wave per (query head, 16-query tile). All operands f16.
// Keys consumed 32 at a time (2 score subtiles), online softmax in registers,
// P reshaped through LDS into A-fragment layout for P*V. Output f16 [S, H*D].
__global__ void __launch_bounds__(32)
flash_attn_kernel(const _Float16* __restrict__ Qb, const _Float16* __restrict__ Kb,
                  const _Float16* __restrict__ Vt, _Float16* __restrict__ Ob) {
  const int qt = blockIdx.x;   // 16-row query tile
  const int hq = blockIdx.y;   // query head
  const int hk = hq >> 2;      // GQA: 4 q heads per kv head

  const _Float16* Q   = Qb + (size_t)(qt * 16) * QSTRIDE + hq * HD;
  const _Float16* Kh  = Kb + hk * HD;                       // ld = KVSTRIDE
  const _Float16* Vth = Vt + (size_t)hk * HD * S_LEN;       // [D][S], ld = S_LEN

  const int lane = threadIdx.x;
  const int hf = lane >> 4, l15 = lane & 15;

  v16h qfrag[4];
#pragma unroll
  for (int c = 0; c < 4; ++c) qfrag[c] = load_a16(Q, QSTRIDE, c * 32);

  v8f oacc[8] = {};
  float mrow[8], lrow[8];
#pragma unroll
  for (int r = 0; r < 8; ++r) { mrow[r] = -1e30f; lrow[r] = 0.0f; }

  __shared__ _Float16 Plds[16 * 32];

  for (int s0 = 0; s0 < S_LEN; s0 += 32) {
    v8f sc0 = {}, sc1 = {};
#pragma unroll
    for (int c = 0; c < 4; ++c) {
      v16h kb0 = load_b16(Kh, KVSTRIDE, s0, c * 32);
      v16h kb1 = load_b16(Kh, KVSTRIDE, s0 + 16, c * 32);
      sc0 = wmma_f16(qfrag[c], kb0, sc0);
      sc1 = wmma_f16(qfrag[c], kb1, sc1);
    }

#pragma unroll
    for (int r = 0; r < 8; ++r) {
      float v = fmaxf(sc0[r], sc1[r]);
#pragma unroll
      for (int msk = 1; msk < 16; msk <<= 1) v = fmaxf(v, __shfl_xor(v, msk, 32));
      const float mnew = fmaxf(mrow[r], v);
      const float p0 = __expf(sc0[r] - mnew);
      const float p1 = __expf(sc1[r] - mnew);
      float psum = p0 + p1;
#pragma unroll
      for (int msk = 1; msk < 16; msk <<= 1) psum += __shfl_xor(psum, msk, 32);
      const float corr = __expf(mrow[r] - mnew);
      lrow[r] = lrow[r] * corr + psum;
      mrow[r] = mnew;
#pragma unroll
      for (int g = 0; g < 8; ++g) oacc[g][r] *= corr;
      Plds[(hf * 8 + r) * 32 + l15]      = (_Float16)p0;
      Plds[(hf * 8 + r) * 32 + 16 + l15] = (_Float16)p1;
    }
    __syncthreads();

    v16h pf;
#pragma unroll
    for (int j = 0; j < 8; ++j) {
      pf[j]     = Plds[l15 * 32 + hf * 8 + j];
      pf[8 + j] = Plds[l15 * 32 + 16 + hf * 8 + j];
    }

#pragma unroll
    for (int g = 0; g < 8; ++g) {
      v16h vb = load_b16(Vth, S_LEN, g * 16, s0);
      oacc[g] = wmma_f16(pf, vb, oacc[g]);
    }
    __syncthreads();
  }

#pragma unroll
  for (int g = 0; g < 8; ++g)
#pragma unroll
    for (int r = 0; r < 8; ++r) {
      const int m = hf * 8 + r;
      Ob[(size_t)(qt * 16 + m) * QSTRIDE + hq * HD + g * 16 + l15] =
          (_Float16)(oacc[g][r] / lrow[r]);
    }
}

extern "C" void kernel_launch(void* const* d_in, const int* in_sizes, int n_in,
                              void* d_out, int out_size, void* d_ws, size_t ws_size,
                              hipStream_t stream) {
  (void)in_sizes; (void)n_in; (void)out_size; (void)ws_size;
  const float* hidden = (const float*)d_in[0];
  const float* freqs  = (const float*)d_in[1];
  const float* Wq     = (const float*)d_in[2];
  const float* Wk     = (const float*)d_in[3];
  const float* Wv     = (const float*)d_in[4];
  const float* Wo     = (const float*)d_in[5];
  const float* qw     = (const float*)d_in[6];
  const float* kw     = (const float*)d_in[7];
  float* out = (float*)d_out;

  // f16 workspace layout (~104 MB)
  _Float16* h16  = (_Float16*)d_ws;
  _Float16* wq16 = h16  + (size_t)S_LEN * HID_DIM;
  _Float16* wk16 = wq16 + (size_t)QSTRIDE * HID_DIM;
  _Float16* wv16 = wk16 + (size_t)KVSTRIDE * HID_DIM;
  _Float16* wo16 = wv16 + (size_t)KVSTRIDE * HID_DIM;
  _Float16* q16  = wo16 + (size_t)HID_DIM * QSTRIDE;
  _Float16* k16  = q16  + (size_t)S_LEN * QSTRIDE;
  _Float16* v16  = k16  + (size_t)S_LEN * KVSTRIDE;
  _Float16* vt16 = v16  + (size_t)S_LEN * KVSTRIDE;
  _Float16* ao16 = vt16 + (size_t)KVSTRIDE * S_LEN;

  const dim3 cb(256);
  const int nh = S_LEN * HID_DIM, nwq = QSTRIDE * HID_DIM;
  const int nwk = KVSTRIDE * HID_DIM, nwo = HID_DIM * QSTRIDE;
  cvt_f16_kernel<<<dim3((nh / 4 + 255) / 256),  cb, 0, stream>>>(hidden, h16, nh);
  cvt_f16_kernel<<<dim3((nwq / 4 + 255) / 256), cb, 0, stream>>>(Wq, wq16, nwq);
  cvt_f16_kernel<<<dim3((nwk / 4 + 255) / 256), cb, 0, stream>>>(Wk, wk16, nwk);
  cvt_f16_kernel<<<dim3((nwk / 4 + 255) / 256), cb, 0, stream>>>(Wv, wv16, nwk);
  cvt_f16_kernel<<<dim3((nwo / 4 + 255) / 256), cb, 0, stream>>>(Wo, wo16, nwo);

  const dim3 gblk(128);
  gemm16_tdm_kernel<_Float16><<<dim3(QSTRIDE / 128, S_LEN / 128), gblk, 0, stream>>>(
      h16, wq16, q16, QSTRIDE, HID_DIM);
  gemm16_tdm_kernel<_Float16><<<dim3(KVSTRIDE / 128, S_LEN / 128), gblk, 0, stream>>>(
      h16, wk16, k16, KVSTRIDE, HID_DIM);
  gemm16_tdm_kernel<_Float16><<<dim3(KVSTRIDE / 128, S_LEN / 128), gblk, 0, stream>>>(
      h16, wv16, v16, KVSTRIDE, HID_DIM);

  norm_rope16_kernel<<<dim3(S_LEN, NH),  dim3(32), 0, stream>>>(
      q16, freqs, qw, QSTRIDE, SOFTMAX_SCALE);
  norm_rope16_kernel<<<dim3(S_LEN, NKV), dim3(32), 0, stream>>>(
      k16, freqs, kw, KVSTRIDE, 1.0f);

  vtrans_kernel<<<dim3(S_LEN / 256, KVSTRIDE), cb, 0, stream>>>(v16, vt16);

  flash_attn_kernel<<<dim3(S_LEN / 16, NH), dim3(32), 0, stream>>>(
      q16, k16, vt16, ao16);

  gemm16_tdm_kernel<float><<<dim3(HID_DIM / 128, S_LEN / 128), gblk, 0, stream>>>(
      ao16, wo16, out, HID_DIM, QSTRIDE);
}
